// NTMModel_87668872446202
// MI455X (gfx1250) — matrix operand order repaired and checked by reading, not compiled, not run on hardware
//
#include <hip/hip_runtime.h>
#include <hip/hip_bf16.h>
#include <math.h>

#define T_STEPS 50
#define BATCH   1024
#define INDIM   405
#define HDIM    200
#define NMEM    128
#define DMEM    40
#define RHEADS  4
#define NWAY    5
#define GAMMA   0.99f

#define KX    416            // IN padded to 13*32
#define KH    224            // H  padded to 7*32
#define KBIG  (KX + KH)      // 640 = 20 k-chunks of 32
#define NGATES 800           // 4*H (50 n-tiles)
#define NKSO   176           // 160 keys + 4 sigma + 5 hWo + 7 pad (11 n-tiles)

typedef __attribute__((ext_vector_type(16))) __bf16 v16bf;
typedef __attribute__((ext_vector_type(8)))  float  v8f;

union Frag { unsigned short u[16]; uint4 q[2]; v16bf v; };

__device__ __forceinline__ unsigned short f2bf(float f) {
    unsigned int u = __float_as_uint(f);
    unsigned int r = (u + 0x7FFFu + ((u >> 16) & 1u)) >> 16;   // RNE
    return (unsigned short)r;
}

// Load a 16x32 bf16 A/B fragment whose two 8-element K-groups are contiguous.
// base points at the start of this row/col's K range; layout per ISA 7.12.2:
//   lanes 0-15  (half 0): K {0..7, 16..23};  lanes 16-31 (half 1): K {8..15, 24..31}
__device__ __forceinline__ v16bf load_fragA(const unsigned short* __restrict__ p,
                                            int kbase, int khalf) {
    Frag f;
    const uint4* a0 = (const uint4*)(p + kbase + 8 * khalf);        // K group lo
    const uint4* a1 = (const uint4*)(p + kbase + 16 + 8 * khalf);   // K group hi
    f.q[0] = *a0; f.q[1] = *a1;
    return f.v;
}
__device__ __forceinline__ v16bf load_fragB(const unsigned short* __restrict__ p,
                                            int kbase, int khalf) {
    Frag f;
    const uint4* b0 = (const uint4*)(p + kbase + 8 * khalf);        // K group lo
    const uint4* b1 = (const uint4*)(p + kbase + 16 + 8 * khalf);   // K group hi
    f.q[0] = *b0; f.q[1] = *b1;
    return f.v;
}

// ---------------- weight packing (once per launch) ----------------

__global__ void prep_wbig_kernel(const float* __restrict__ W_ih,
                                 const float* __restrict__ W_hh,
                                 unsigned short* __restrict__ Wbig) {
    int i = blockIdx.x * blockDim.x + threadIdx.x;
    if (i >= NGATES * KBIG) return;
    int n = i / KBIG, k = i % KBIG;
    float v = 0.f;
    if (k < KX) { if (k < INDIM) v = W_ih[(size_t)n * INDIM + k]; }
    else        { int kh = k - KX; if (kh < HDIM) v = W_hh[(size_t)n * HDIM + kh]; }
    Wbig[i] = f2bf(v);
}

__global__ void prep_wkso_kernel(const float* __restrict__ Wk,
                                 const float* __restrict__ Ws,
                                 const float* __restrict__ Wo,
                                 unsigned short* __restrict__ Wkso) {
    int i = blockIdx.x * blockDim.x + threadIdx.x;
    if (i >= NKSO * KH) return;
    int n = i / KH, k = i % KH;
    float v = 0.f;
    if (k < HDIM) {
        if      (n < RHEADS * DMEM)          v = Wk[(size_t)n * HDIM + k];
        else if (n < RHEADS * DMEM + RHEADS) v = Ws[(size_t)(n - 160) * HDIM + k];
        else if (n < RHEADS * DMEM + RHEADS + NWAY)
            v = Wo[(size_t)(n - 164) * (HDIM + RHEADS * DMEM) + k];  // Wo[:, :200]
    }
    Wkso[i] = f2bf(v);
}

__global__ void fill_kernel(float* __restrict__ p, float v, int n) {
    int i = blockIdx.x * blockDim.x + threadIdx.x;
    if (i < n) p[i] = v;
}

// ---------------- per-step x_t -> bf16 into Abuf cols [0,405) ----------------
// Abuf is [BATCH, KBIG] bf16; pad cols stay zero from init.

__global__ void convert_x_kernel(const float* __restrict__ x_t,
                                 unsigned short* __restrict__ Abuf) {
    int i = blockIdx.x * blockDim.x + threadIdx.x;
    if (i >= BATCH * INDIM) return;
    int b = i / INDIM, k = i % INDIM;
    Abuf[(size_t)b * KBIG + k] = f2bf(x_t[i]);
}

// ---------------- gates GEMM: [1024 x 800] = Abuf @ Wbig^T ----------------
// Block = 128 thr (4 waves); block tile 128x32; wave tile 32x32 (2M x 2N).

__global__ __launch_bounds__(128) void gemm_gates_kernel(
        const unsigned short* __restrict__ Abuf,   // [BATCH, KBIG] bf16
        const unsigned short* __restrict__ Wbig,   // [NGATES, KBIG] bf16
        float* __restrict__ gates) {               // [BATCH, NGATES]
    const int wave  = threadIdx.x >> 5;
    const int lane  = threadIdx.x & 31;
    const int ln16  = lane & 15;
    const int khalf = lane >> 4;

    const int bn = blockIdx.x % (NGATES / 32);     // 25
    const int bm = blockIdx.x / (NGATES / 32);     // 8
    const int rowBase = bm * 128 + wave * 32;
    const int colBase = bn * 32;

    const unsigned short* a0p = Abuf + (size_t)(rowBase + ln16) * KBIG;
    const unsigned short* a1p = Abuf + (size_t)(rowBase + 16 + ln16) * KBIG;
    const unsigned short* b0p = Wbig + (size_t)(colBase + ln16) * KBIG;
    const unsigned short* b1p = Wbig + (size_t)(colBase + 16 + ln16) * KBIG;

    v8f acc00 = {}, acc01 = {}, acc10 = {}, acc11 = {};
    for (int kt = 0; kt < KBIG / 32; ++kt) {
        const int kb = kt * 32;
        v16bf a0 = load_fragA(a0p, kb, khalf);
        v16bf a1 = load_fragA(a1p, kb, khalf);
        v16bf b0 = load_fragB(b0p, kb, khalf);
        v16bf b1 = load_fragB(b1p, kb, khalf);
        acc00 = __builtin_amdgcn_wmma_f32_16x16x32_bf16(false, a0, false, b0, (short)0, acc00, false, false);
        acc01 = __builtin_amdgcn_wmma_f32_16x16x32_bf16(false, a0, false, b1, (short)0, acc01, false, false);
        acc10 = __builtin_amdgcn_wmma_f32_16x16x32_bf16(false, a1, false, b0, (short)0, acc10, false, false);
        acc11 = __builtin_amdgcn_wmma_f32_16x16x32_bf16(false, a1, false, b1, (short)0, acc11, false, false);
    }
#pragma unroll
    for (int v = 0; v < 8; ++v) {
        int r0 = rowBase + v + 8 * khalf;          // D: M = vgpr + 8*lanehalf
        int r1 = r0 + 16;
        gates[(size_t)r0 * NGATES + colBase + ln16]      = acc00[v];
        gates[(size_t)r0 * NGATES + colBase + 16 + ln16] = acc01[v];
        gates[(size_t)r1 * NGATES + colBase + ln16]      = acc10[v];
        gates[(size_t)r1 * NGATES + colBase + 16 + ln16] = acc11[v];
    }
}

// ---------------- heads GEMM: [1024 x 176] = h(bf16) @ Wkso^T ----------------
// A read from Abuf h-region (col offset KX, stride KBIG). Wave tile 32x16.

__global__ __launch_bounds__(128) void gemm_kso_kernel(
        const unsigned short* __restrict__ Abuf,   // [BATCH, KBIG] bf16 (h at +KX)
        const unsigned short* __restrict__ Wkso,   // [NKSO, KH] bf16
        float* __restrict__ kso) {                 // [BATCH, NKSO]
    const int wave  = threadIdx.x >> 5;
    const int lane  = threadIdx.x & 31;
    const int ln16  = lane & 15;
    const int khalf = lane >> 4;

    const int bn = blockIdx.x % (NKSO / 16);       // 11
    const int bm = blockIdx.x / (NKSO / 16);       // 8
    const int rowBase = bm * 128 + wave * 32;
    const int col = bn * 16 + ln16;

    const unsigned short* a0p = Abuf + (size_t)(rowBase + ln16) * KBIG + KX;
    const unsigned short* a1p = Abuf + (size_t)(rowBase + 16 + ln16) * KBIG + KX;
    const unsigned short* bp  = Wkso + (size_t)col * KH;

    v8f acc0 = {}, acc1 = {};
    for (int kt = 0; kt < KH / 32; ++kt) {
        const int kb = kt * 32;
        v16bf a0 = load_fragA(a0p, kb, khalf);
        v16bf a1 = load_fragA(a1p, kb, khalf);
        v16bf b  = load_fragB(bp, kb, khalf);
        acc0 = __builtin_amdgcn_wmma_f32_16x16x32_bf16(false, a0, false, b, (short)0, acc0, false, false);
        acc1 = __builtin_amdgcn_wmma_f32_16x16x32_bf16(false, a1, false, b, (short)0, acc1, false, false);
    }
#pragma unroll
    for (int v = 0; v < 8; ++v) {
        int r0 = rowBase + v + 8 * khalf;
        kso[(size_t)r0 * NKSO + col]        = acc0[v];
        kso[(size_t)(r0 + 16) * NKSO + col] = acc1[v];
    }
}

// ---------------- LSTM elementwise; writes h as bf16 into Abuf h-region ----------------

__global__ void lstm_kernel(const float* __restrict__ gates,
                            const float* __restrict__ b_ih,
                            const float* __restrict__ b_hh,
                            float* __restrict__ c,
                            unsigned short* __restrict__ Abuf) {
    int i = blockIdx.x * blockDim.x + threadIdx.x;
    if (i >= BATCH * HDIM) return;
    int b = i / HDIM, j = i % HDIM;
    const float* g = gates + (size_t)b * NGATES;
    float ig = g[j]            + b_ih[j]            + b_hh[j];
    float fg = g[HDIM + j]     + b_ih[HDIM + j]     + b_hh[HDIM + j];
    float gg = g[2 * HDIM + j] + b_ih[2 * HDIM + j] + b_hh[2 * HDIM + j];
    float og = g[3 * HDIM + j] + b_ih[3 * HDIM + j] + b_hh[3 * HDIM + j];
    float si = 1.f / (1.f + expf(-ig));
    float sf = 1.f / (1.f + expf(-fg));
    float so = 1.f / (1.f + expf(-og));
    float cn = sf * c[i] + si * tanhf(gg);
    c[i] = cn;
    float hv = so * tanhf(cn);
    Abuf[(size_t)b * KBIG + KX + j] = f2bf(hv);
}

// ---------------- NTM memory module: one block (128 thr) per batch elem ----------------

__global__ __launch_bounds__(128) void mem_step_kernel(
        const float* __restrict__ kso,   // [B, NKSO]: keys(160) | sigma(4) | hWo(5)
        const float* __restrict__ bk, const float* __restrict__ bs,
        const float* __restrict__ bo, const float* __restrict__ Wo,  // [NWAY, 360]
        float* __restrict__ M, float* __restrict__ wr,
        float* __restrict__ wu_g, float* __restrict__ wlu_g,
        float* __restrict__ out_t) {     // [B, NWAY]
    const int b = blockIdx.x;
    const int tid = threadIdx.x;         // 0..127, owns memory slot `tid`

    __shared__ float sM[NMEM][DMEM + 1];
    __shared__ float sk[RHEADS][DMEM];
    __shared__ float ssig[RHEADS];
    __shared__ float snk[RHEADS];
    __shared__ float swr[RHEADS][NMEM];  // wr in, overwritten with wr_new
    __shared__ float sww[RHEADS][NMEM];
    __shared__ float srout[RHEADS * DMEM];
    __shared__ float red[NMEM];
    __shared__ int   redi[NMEM];
    __shared__ float slog[NWAY];

    float* Mb = M + (size_t)b * NMEM * DMEM;
    const float* ks = kso + (size_t)b * NKSO;

    for (int d = 0; d < DMEM; ++d) sM[tid][d] = Mb[tid * DMEM + d];
    for (int i = tid; i < RHEADS * DMEM; i += 128)
        sk[i / DMEM][i % DMEM] = tanhf(ks[i] + bk[i]);
    if (tid < RHEADS)
        ssig[tid] = 1.f / (1.f + expf(-tanhf(ks[160 + tid] + bs[tid])));
    for (int i = tid; i < RHEADS * NMEM; i += 128)
        swr[i / NMEM][i % NMEM] = wr[(size_t)b * RHEADS * NMEM + i];
    float wuv  = wu_g[(size_t)b * NMEM + tid];
    float wluv = wlu_g[(size_t)b * NMEM + tid];
    __syncthreads();

    // argmin(wu) -> zero that memory row (first occurrence on ties)
    red[tid] = wuv; redi[tid] = tid;
    __syncthreads();
    for (int s = 64; s > 0; s >>= 1) {
        if (tid < s) {
            float v2 = red[tid + s]; int i2 = redi[tid + s];
            if (v2 < red[tid] || (v2 == red[tid] && i2 < redi[tid])) { red[tid] = v2; redi[tid] = i2; }
        }
        __syncthreads();
    }
    int idxmin = redi[0];
    __syncthreads();
    if (tid < DMEM) sM[idxmin][tid] = 0.f;
    __syncthreads();

    // write weights ww and memory write M += ww^T k
    float wwv[RHEADS];
#pragma unroll
    for (int r = 0; r < RHEADS; ++r) {
        float w = ssig[r] * swr[r][tid] + (1.f - ssig[r]) * wluv;
        wwv[r] = w; sww[r][tid] = w;
    }
    float nm = 0.f;
    for (int d = 0; d < DMEM; ++d) {
        float a = sM[tid][d];
#pragma unroll
        for (int r = 0; r < RHEADS; ++r) a += wwv[r] * sk[r][d];
        sM[tid][d] = a;
        nm += a * a;
    }
    nm = sqrtf(nm);
    if (tid < RHEADS) {
        float s2 = 0.f;
        for (int d = 0; d < DMEM; ++d) { float v = sk[tid][d]; s2 += v * v; }
        snk[tid] = sqrtf(s2);
    }
    __syncthreads();

    // cosine similarity + softmax over memory slots per read head
    float Kv[RHEADS];
#pragma unroll
    for (int r = 0; r < RHEADS; ++r) {
        float dot = 0.f;
        for (int d = 0; d < DMEM; ++d) dot += sk[r][d] * sM[tid][d];
        Kv[r] = dot / (snk[r] * nm);
    }
    for (int r = 0; r < RHEADS; ++r) {
        __syncthreads();
        red[tid] = Kv[r];
        __syncthreads();
        for (int s = 64; s > 0; s >>= 1) {
            if (tid < s) { float v2 = red[tid + s]; if (v2 > red[tid]) red[tid] = v2; }
            __syncthreads();
        }
        float mx = red[0];
        __syncthreads();
        float e = expf(Kv[r] - mx);
        red[tid] = e;
        __syncthreads();
        for (int s = 64; s > 0; s >>= 1) {
            if (tid < s) red[tid] += red[tid + s];
            __syncthreads();
        }
        swr[r][tid] = e / red[0];          // wr_new
    }
    __syncthreads();

    // read vectors r_out[r,d] = sum_m wr_new[r,m] * M[m,d]
    for (int i = tid; i < RHEADS * DMEM; i += 128) {
        int r = i / DMEM, d = i % DMEM;
        float a = 0.f;
        for (int m = 0; m < NMEM; ++m) a += swr[r][m] * sM[m][d];
        srout[i] = a;
    }
    __syncthreads();

    // output head: logits = hWo + r_out @ Wo[:,200:]^T + bo -> sigmoid -> softmax
    if (tid < NWAY) {
        float lg = ks[164 + tid] + bo[tid];
        const float* wrow = Wo + (size_t)tid * (HDIM + RHEADS * DMEM) + HDIM;
        for (int i = 0; i < RHEADS * DMEM; ++i) lg += srout[i] * wrow[i];
        slog[tid] = 1.f / (1.f + expf(-lg));
    }
    __syncthreads();
    if (tid == 0) {
        float mx = slog[0];
        for (int w = 1; w < NWAY; ++w) mx = fmaxf(mx, slog[w]);
        float es[NWAY]; float s2 = 0.f;
        for (int w = 0; w < NWAY; ++w) { es[w] = expf(slog[w] - mx); s2 += es[w]; }
        for (int w = 0; w < NWAY; ++w) out_t[(size_t)b * NWAY + w] = es[w] / s2;
    }

    // usage update + 4th smallest (R rounds of min-with-exclusion)
    float wun = GAMMA * wuv;
#pragma unroll
    for (int r = 0; r < RHEADS; ++r) wun += swr[r][tid] + sww[r][tid];
    float myv = wun;
    float kth = 0.f;
    for (int rr = 0; rr < RHEADS; ++rr) {
        __syncthreads();
        red[tid] = myv; redi[tid] = tid;
        __syncthreads();
        for (int s = 64; s > 0; s >>= 1) {
            if (tid < s) {
                float v2 = red[tid + s]; int i2 = redi[tid + s];
                if (v2 < red[tid] || (v2 == red[tid] && i2 < redi[tid])) { red[tid] = v2; redi[tid] = i2; }
            }
            __syncthreads();
        }
        kth = red[0];
        if (tid == redi[0]) myv = __builtin_inff();
    }
    float wlun = (wun <= kth) ? 1.f : 0.f;

    // state write-back
    wu_g [(size_t)b * NMEM + tid] = wun;
    wlu_g[(size_t)b * NMEM + tid] = wlun;
#pragma unroll
    for (int r = 0; r < RHEADS; ++r)
        wr[(size_t)b * RHEADS * NMEM + r * NMEM + tid] = swr[r][tid];
    for (int d = 0; d < DMEM; ++d) Mb[tid * DMEM + d] = sM[tid][d];
}

// ---------------- launch ----------------

extern "C" void kernel_launch(void* const* d_in, const int* in_sizes, int n_in,
                              void* d_out, int out_size, void* d_ws, size_t ws_size,
                              hipStream_t stream) {
    const float* x    = (const float*)d_in[0];
    const float* W_ih = (const float*)d_in[1];
    const float* W_hh = (const float*)d_in[2];
    const float* b_ih = (const float*)d_in[3];
    const float* b_hh = (const float*)d_in[4];
    const float* Wk   = (const float*)d_in[5];
    const float* bk   = (const float*)d_in[6];
    const float* Wsm  = (const float*)d_in[7];
    const float* bs   = (const float*)d_in[8];
    const float* Wo   = (const float*)d_in[9];
    const float* bo   = (const float*)d_in[10];
    float* out = (float*)d_out;

    char* wsb = (char*)d_ws;
    size_t off = 0;
    auto alloc = [&](size_t bytes) -> char* {
        char* p = wsb + off;
        off += (bytes + 255) & ~(size_t)255;
        return p;
    };
    unsigned short* Wbig = (unsigned short*)alloc((size_t)NGATES * KBIG * 2);
    unsigned short* Wkso = (unsigned short*)alloc((size_t)NKSO * KH * 2);
    unsigned short* Abuf = (unsigned short*)alloc((size_t)BATCH * KBIG * 2);
    float* gates  = (float*)alloc((size_t)BATCH * NGATES * 4);
    float* cbuf   = (float*)alloc((size_t)BATCH * HDIM * 4);
    float* ksob   = (float*)alloc((size_t)BATCH * NKSO * 4);
    float* Mbuf   = (float*)alloc((size_t)BATCH * NMEM * DMEM * 4);
    float* wrbuf  = (float*)alloc((size_t)BATCH * RHEADS * NMEM * 4);
    float* wubuf  = (float*)alloc((size_t)BATCH * NMEM * 4);
    float* wlubuf = (float*)alloc((size_t)BATCH * NMEM * 4);

    // pack weights to bf16 (deterministic, re-done every call)
    prep_wbig_kernel<<<(NGATES * KBIG + 255) / 256, 256, 0, stream>>>(W_ih, W_hh, Wbig);
    prep_wkso_kernel<<<(NKSO * KH + 255) / 256, 256, 0, stream>>>(Wk, Wsm, Wo, Wkso);

    // init state; Abuf zeroed so pad columns (405..415, 616..639) and h start as 0
    fill_kernel<<<(BATCH * KBIG / 2 + 255) / 256, 256, 0, stream>>>((float*)Abuf, 0.f, BATCH * KBIG / 2);
    fill_kernel<<<(BATCH * HDIM + 255) / 256, 256, 0, stream>>>(cbuf, 0.f, BATCH * HDIM);
    fill_kernel<<<(BATCH * NMEM * DMEM + 255) / 256, 256, 0, stream>>>(Mbuf, 0.f, BATCH * NMEM * DMEM);
    fill_kernel<<<(BATCH * RHEADS * NMEM + 255) / 256, 256, 0, stream>>>(wrbuf, 0.f, BATCH * RHEADS * NMEM);
    fill_kernel<<<(BATCH * NMEM + 255) / 256, 256, 0, stream>>>(wubuf, 0.f, BATCH * NMEM);
    fill_kernel<<<(BATCH * NMEM + 255) / 256, 256, 0, stream>>>(wlubuf, 1.f, BATCH * NMEM);

    for (int t = 0; t < T_STEPS; ++t) {
        convert_x_kernel<<<(BATCH * INDIM + 255) / 256, 256, 0, stream>>>(
            x + (size_t)t * BATCH * INDIM, Abuf);
        gemm_gates_kernel<<<(BATCH / 128) * (NGATES / 32), 128, 0, stream>>>(
            Abuf, Wbig, gates);
        lstm_kernel<<<(BATCH * HDIM + 255) / 256, 256, 0, stream>>>(
            gates, b_ih, b_hh, cbuf, Abuf);
        gemm_kso_kernel<<<(BATCH / 128) * (NKSO / 16), 128, 0, stream>>>(
            Abuf, Wkso, ksob);
        mem_step_kernel<<<BATCH, 128, 0, stream>>>(
            ksob, bk, bs, bo, Wo, Mbuf, wrbuf, wubuf, wlubuf,
            out + (size_t)t * BATCH * NWAY);
    }
}
